// MultiHeadAttention_44659069944106
// MI455X (gfx1250) — matrix operand order repaired and checked by reading, not compiled, not run on hardware
//
#include <hip/hip_runtime.h>
#include <hip/hip_bf16.h>
#include <math.h>

// ---------------------------------------------------------------------------
// MHA forward for B=4, L=2048, D=768, H=12, d=64 on gfx1250 (CDNA5, wave32).
// All GEMMs via v_wmma_f32_16x16x32_bf16 (f32 accum). Flash-attention inner
// loop; every WMMA fragment is fed by contiguous ds_load_b128. Straight-copy
// tiles (GEMM A, attention K) are staged with the async global->LDS engine
// (ASYNCcnt); transposing tiles use the register path.
// ---------------------------------------------------------------------------

typedef __bf16 v16bf __attribute__((ext_vector_type(16)));
typedef __bf16 v8bf  __attribute__((ext_vector_type(8)));
typedef float  v8f   __attribute__((ext_vector_type(8)));
typedef int    v4i   __attribute__((ext_vector_type(4)));

#define BATCH 4
#define SEQ   2048
#define DMODEL 768
#define NHEAD 12
#define HDIM  64
#define MTOT  (BATCH * SEQ)       // 8192

#define LDA 40                    // padded LDS stride (elems): 20 dwords, bank-clean
#define LDK 72                    // padded stride for K tile rows (36 dwords)

#if defined(__has_builtin)
#if __has_builtin(__builtin_amdgcn_global_load_async_to_lds_b128)
#define USE_ASYNC_LDS 1
#endif
#endif

#ifdef USE_ASYNC_LDS
typedef __attribute__((address_space(1))) v4i* g_v4i_p;
typedef __attribute__((address_space(3))) v4i* l_v4i_p;
// Generic LDS pointers carry the LDS offset in their low 32 bits (aperture
// mapping truncates); route through integers to sidestep AS-cast diagnostics.
__device__ inline g_v4i_p as_glb(const void* p) {
  return (g_v4i_p)(uintptr_t)p;
}
__device__ inline l_v4i_p as_lds(const void* p) {
  return (l_v4i_p)(uintptr_t)(uint32_t)(uintptr_t)p;
}
#define ASYNC_COPY16(dst, src) \
  __builtin_amdgcn_global_load_async_to_lds_b128(as_glb(src), as_lds(dst), 0, 0)
#define ASYNC_DRAIN() asm volatile("s_wait_asynccnt 0x0" ::: "memory")
#else
#define ASYNC_COPY16(dst, src) \
  (*reinterpret_cast<uint4*>(dst) = *reinterpret_cast<const uint4*>(src))
#define ASYNC_DRAIN() ((void)0)
#endif

// ---- WMMA fragment loaders (contiguous k = 16*(lane>>4) + i order, wave32) --------------

// A-matrix 16x32 bf16 from row-major LDS/global (leading dim ld).
// lane<16: M=lane, K=[0..15]; lane>=16: M=lane-16, K=[16..31]
__device__ inline v16bf load_a_frag(const __bf16* __restrict__ base, int ld) {
  int lane = threadIdx.x & 31;
  const __bf16* p = base + (size_t)(lane & 15) * ld + ((lane >> 4) << 4);
  v16bf a;
#pragma unroll
  for (int i = 0; i < 16; ++i) a[i] = p[i];
  return a;
}

// B-matrix 32x16 bf16 fragment where the source holds B *transposed*
// (N-major rows, K contiguous): lane N = lane&15, K = (lane>>4)*16 + [0..15]
// -> 16 contiguous elements (two ds_load_b128).
__device__ inline v16bf load_b_frag_contig(const __bf16* __restrict__ p) {
  v16bf b;
#pragma unroll
  for (int i = 0; i < 16; ++i) b[i] = p[i];
  return b;
}

__device__ inline v8f wmma_bf16(v16bf a, v16bf b, v8f c) {
  return __builtin_amdgcn_wmma_f32_16x16x32_bf16(false, a, false, b,
                                                 (short)0, c, false, false);
}

// ---- fp32 -> bf16 cast kernel ---------------------------------------------
__global__ __launch_bounds__(256) void cast_bf16_kernel(const float* __restrict__ in,
                                                        __bf16* __restrict__ out, int n) {
  int i = blockIdx.x * 256 + threadIdx.x;
  if (i < n) out[i] = (__bf16)in[i];
}

// ---------------------------------------------------------------------------
// Shared GEMM body: 128x128 WG tile, 8 waves each owning 64x32, K-step 32.
// A staged row-major (padded) via async copy, B staged TRANSPOSED.
// ---------------------------------------------------------------------------
__device__ inline void gemm_stage(const __bf16* __restrict__ A,
                                  const __bf16* __restrict__ Bw,
                                  __bf16* __restrict__ As, __bf16* __restrict__ BsT,
                                  int tileM, int tileN, int kt, int Kk, int Nn) {
  const int t = threadIdx.x;
#pragma unroll
  for (int p = 0; p < 2; ++p) {                 // A: 128x32, 512 16B units
    int uid = t + p * 256;
    int row = uid >> 2, seg = uid & 3;
    ASYNC_COPY16(As + (size_t)row * LDA + seg * 8,
                 A + (size_t)(tileM + row) * Kk + kt + seg * 8);
  }
#pragma unroll
  for (int p = 0; p < 2; ++p) {                 // B: 32x128 -> transpose into BsT
    int uid = t + p * 256;
    int k = uid >> 4, seg = uid & 15;
    v8bf v = *reinterpret_cast<const v8bf*>(Bw + (size_t)(kt + k) * Nn + tileN + seg * 8);
#pragma unroll
    for (int j = 0; j < 8; ++j)
      BsT[(size_t)(seg * 8 + j) * LDA + k] = v[j];
  }
  ASYNC_DRAIN();
}

__device__ inline void gemm_compute(const __bf16* __restrict__ As,
                                    const __bf16* __restrict__ BsT,
                                    int wm, int wn, int lane, v8f acc[4][2]) {
  v16bf af[4], bf_[2];
#pragma unroll
  for (int sm = 0; sm < 4; ++sm)
    af[sm] = load_a_frag(As + (size_t)(wm * 64 + sm * 16) * LDA, LDA);
#pragma unroll
  for (int sn = 0; sn < 2; ++sn)
    bf_[sn] = load_b_frag_contig(
        BsT + (size_t)(wn * 32 + sn * 16 + (lane & 15)) * LDA + ((lane >> 4) << 4));
#pragma unroll
  for (int sm = 0; sm < 4; ++sm)
#pragma unroll
    for (int sn = 0; sn < 2; ++sn)
      acc[sm][sn] = wmma_bf16(af[sm], bf_[sn], acc[sm][sn]);
}

// ---- QKV projection: [8192x768] @ [768x2304] + bias -> Q/K/V [B,H,L,64] ----
__global__ __launch_bounds__(256) void gemm_qkv_kernel(
    const __bf16* __restrict__ A, const __bf16* __restrict__ Bw,
    const float* __restrict__ bias,
    __bf16* __restrict__ Q, __bf16* __restrict__ K, __bf16* __restrict__ V) {
  const int Kk = DMODEL, Nn = 3 * DMODEL;
  __shared__ __align__(16) __bf16 As[128 * LDA];
  __shared__ __align__(16) __bf16 BsT[128 * LDA];

  const int t = threadIdx.x, wave = t >> 5, lane = t & 31;
  const int wm = wave >> 2, wn = wave & 3;
  const int tileM = blockIdx.y * 128, tileN = blockIdx.x * 128;

  v8f acc[4][2];
#pragma unroll
  for (int i = 0; i < 4; ++i)
#pragma unroll
    for (int j = 0; j < 2; ++j) acc[i][j] = (v8f)(0.0f);

  for (int kt = 0; kt < Kk; kt += 32) {
    __syncthreads();
    gemm_stage(A, Bw, As, BsT, tileM, tileN, kt, Kk, Nn);
    __syncthreads();
    gemm_compute(As, BsT, wm, wn, lane, acc);
  }

#pragma unroll
  for (int sm = 0; sm < 4; ++sm)
#pragma unroll
    for (int sn = 0; sn < 2; ++sn)
#pragma unroll
      for (int r = 0; r < 8; ++r) {
        int grow = tileM + wm * 64 + sm * 16 + ((lane >> 4) << 3) + r;   // b*L + l
        int gcol = tileN + wn * 32 + sn * 16 + (lane & 15);              // 0..2303
        float val = acc[sm][sn][r] + bias[gcol];
        int which = gcol / DMODEL;
        int rem = gcol - which * DMODEL;
        int h = rem >> 6, d = rem & 63;
        int b = grow >> 11, l = grow & 2047;
        size_t o = (((size_t)(b * NHEAD + h)) * SEQ + l) * HDIM + d;
        __bf16 bv = (__bf16)val;
        if (which == 0) Q[o] = bv;
        else if (which == 1) K[o] = bv;
        else V[o] = bv;
      }
}

// ---- Flash attention: one block per (b, h, 128-query tile) -----------------
__global__ __launch_bounds__(256) void attn_kernel(
    const __bf16* __restrict__ Q, const __bf16* __restrict__ Kg,
    const __bf16* __restrict__ Vg, __bf16* __restrict__ Aout) {
  __shared__ __align__(16) __bf16 Ks[32 * LDK];       // [key][d], padded rows
  __shared__ __align__(16) __bf16 VsT[64 * LDA];      // [d][key], K contiguous
  __shared__ __align__(16) __bf16 Ps[8][16 * LDA];    // per-wave P patch

  const int t = threadIdx.x, wave = t >> 5, lane = t & 31;
  const int qtile = blockIdx.x, h = blockIdx.y, b = blockIdx.z;
  const size_t bh = ((size_t)(b * NHEAD + h)) * SEQ * HDIM;
  const int qrow0 = qtile * 128 + wave * 16;

  // Q fragments held in registers for the whole pass (d=64 -> two K-chunks)
  const __bf16* qbase = Q + bh + (size_t)qrow0 * HDIM;
  v16bf qa0 = load_a_frag(qbase, HDIM);
  v16bf qa1 = load_a_frag(qbase + 32, HDIM);

  v8f o[4];
#pragma unroll
  for (int n = 0; n < 4; ++n) o[n] = (v8f)(0.0f);
  float mrow[8], lrow[8];
#pragma unroll
  for (int r = 0; r < 8; ++r) { mrow[r] = -3.0e30f; lrow[r] = 0.0f; }

  for (int kt = 0; kt < SEQ / 32; ++kt) {
    __syncthreads();
    {   // stage K (async straight copy) and V (register transpose) 32x64 tiles
      int key = t >> 3, seg = t & 7;
      size_t goff = bh + (size_t)(kt * 32 + key) * HDIM + seg * 8;
      ASYNC_COPY16(Ks + (size_t)key * LDK + seg * 8, Kg + goff);
      v8bf vv = *reinterpret_cast<const v8bf*>(Vg + goff);
#pragma unroll
      for (int j = 0; j < 8; ++j)
        VsT[(size_t)(seg * 8 + j) * LDA + key] = vv[j];
    }
    ASYNC_DRAIN();
    __syncthreads();

    // S = Q (16x64) @ K^T (64x32): K^T's B-frag is contiguous in Ks[key][d]
    v8f s[2];
    s[0] = (v8f)(0.0f); s[1] = (v8f)(0.0f);
#pragma unroll
    for (int sn = 0; sn < 2; ++sn) {
      int key = sn * 16 + (lane & 15);
      const __bf16* kb = Ks + (size_t)key * LDK + ((lane >> 4) << 4);
      s[sn] = wmma_bf16(qa0, load_b_frag_contig(kb), s[sn]);
      s[sn] = wmma_bf16(qa1, load_b_frag_contig(kb + 32), s[sn]);
    }

    // online softmax; a row spans one 16-lane half of the wave
    float pv[2][8], rmax[8], rsum[8];
#pragma unroll
    for (int r = 0; r < 8; ++r) {
      float a = s[0][r] * 0.125f, c = s[1][r] * 0.125f;   // 1/sqrt(64)
      s[0][r] = a; s[1][r] = c;
      rmax[r] = fmaxf(a, c);
    }
#pragma unroll
    for (int off = 1; off < 16; off <<= 1)
#pragma unroll
      for (int r = 0; r < 8; ++r)
        rmax[r] = fmaxf(rmax[r], __shfl_xor(rmax[r], off, 32));
#pragma unroll
    for (int r = 0; r < 8; ++r) {
      float nm = fmaxf(mrow[r], rmax[r]);
      float alpha = __expf(mrow[r] - nm);
      mrow[r] = nm;
      pv[0][r] = __expf(s[0][r] - nm);
      pv[1][r] = __expf(s[1][r] - nm);
      rsum[r] = pv[0][r] + pv[1][r];
      lrow[r] *= alpha;
#pragma unroll
      for (int n = 0; n < 4; ++n) o[n][r] *= alpha;
    }
#pragma unroll
    for (int off = 1; off < 16; off <<= 1)
#pragma unroll
      for (int r = 0; r < 8; ++r)
        rsum[r] += __shfl_xor(rsum[r], off, 32);
#pragma unroll
    for (int r = 0; r < 8; ++r) lrow[r] += rsum[r];

    // round-trip P through the per-wave LDS patch into A-fragment layout
#pragma unroll
    for (int sn = 0; sn < 2; ++sn)
#pragma unroll
      for (int r = 0; r < 8; ++r)
        Ps[wave][(size_t)(((lane >> 4) << 3) + r) * LDA + sn * 16 + (lane & 15)] =
            (__bf16)pv[sn][r];
    asm volatile("s_wait_dscnt 0" ::: "memory");
    v16bf pa = load_a_frag(&Ps[wave][0], LDA);

    // O += P (16x32) @ V (32x64): V's B-frag contiguous in VsT[d][key]
#pragma unroll
    for (int n = 0; n < 4; ++n) {
      v16bf vf = load_b_frag_contig(
          VsT + (size_t)(n * 16 + (lane & 15)) * LDA + ((lane >> 4) << 4));
      o[n] = wmma_bf16(pa, vf, o[n]);
    }
  }

  // write attention output as bf16 [B*L, 768] for the final projection
#pragma unroll
  for (int n = 0; n < 4; ++n)
#pragma unroll
    for (int r = 0; r < 8; ++r) {
      int row = qrow0 + ((lane >> 4) << 3) + r;
      int d = n * 16 + (lane & 15);
      float val = o[n][r] / lrow[r];
      Aout[((size_t)(b * SEQ + row)) * DMODEL + h * HDIM + d] = (__bf16)val;
    }
}

// ---- Output projection: [8192x768] @ [768x768] + bias -> f32 ---------------
__global__ __launch_bounds__(256) void gemm_out_kernel(
    const __bf16* __restrict__ A, const __bf16* __restrict__ Bw,
    const float* __restrict__ bias, float* __restrict__ C) {
  const int Kk = DMODEL, Nn = DMODEL;
  __shared__ __align__(16) __bf16 As[128 * LDA];
  __shared__ __align__(16) __bf16 BsT[128 * LDA];

  const int t = threadIdx.x, wave = t >> 5, lane = t & 31;
  const int wm = wave >> 2, wn = wave & 3;
  const int tileM = blockIdx.y * 128, tileN = blockIdx.x * 128;

  v8f acc[4][2];
#pragma unroll
  for (int i = 0; i < 4; ++i)
#pragma unroll
    for (int j = 0; j < 2; ++j) acc[i][j] = (v8f)(0.0f);

  for (int kt = 0; kt < Kk; kt += 32) {
    __syncthreads();
    gemm_stage(A, Bw, As, BsT, tileM, tileN, kt, Kk, Nn);
    __syncthreads();
    gemm_compute(As, BsT, wm, wn, lane, acc);
  }

#pragma unroll
  for (int sm = 0; sm < 4; ++sm)
#pragma unroll
    for (int sn = 0; sn < 2; ++sn)
#pragma unroll
      for (int r = 0; r < 8; ++r) {
        int grow = tileM + wm * 64 + sm * 16 + ((lane >> 4) << 3) + r;
        int gcol = tileN + wn * 32 + sn * 16 + (lane & 15);
        C[(size_t)grow * Nn + gcol] = acc[sm][sn][r] + bias[gcol];
      }
}

// ---------------------------------------------------------------------------
extern "C" void kernel_launch(void* const* d_in, const int* in_sizes, int n_in,
                              void* d_out, int out_size, void* d_ws, size_t ws_size,
                              hipStream_t stream) {
  const float* x     = (const float*)d_in[0];   // [4,2048,768]
  const float* W_qkv = (const float*)d_in[1];   // [768,2304]
  const float* b_qkv = (const float*)d_in[2];   // [2304]
  const float* W_out = (const float*)d_in[3];   // [768,768]
  const float* b_out = (const float*)d_in[4];   // [768]
  float* out = (float*)d_out;

  char* ws = (char*)d_ws;
  const size_t nX    = (size_t)MTOT * DMODEL;
  const size_t nWqkv = (size_t)DMODEL * 3 * DMODEL;
  const size_t nWout = (size_t)DMODEL * DMODEL;
  const size_t nQKV  = (size_t)MTOT * DMODEL;

  __bf16* xb    = (__bf16*)ws;                 ws += nX * 2;
  __bf16* wqkvb = (__bf16*)ws;                 ws += nWqkv * 2;
  __bf16* woutb = (__bf16*)ws;                 ws += nWout * 2;
  __bf16* qb    = (__bf16*)ws;                 ws += nQKV * 2;
  __bf16* kb    = (__bf16*)ws;                 ws += nQKV * 2;
  __bf16* vb    = (__bf16*)ws;                 ws += nQKV * 2;
  __bf16* attnb = (__bf16*)ws;                 ws += nX * 2;

  cast_bf16_kernel<<<dim3((nX + 255) / 256), 256, 0, stream>>>(x, xb, (int)nX);
  cast_bf16_kernel<<<dim3((nWqkv + 255) / 256), 256, 0, stream>>>(W_qkv, wqkvb, (int)nWqkv);
  cast_bf16_kernel<<<dim3((nWout + 255) / 256), 256, 0, stream>>>(W_out, woutb, (int)nWout);

  gemm_qkv_kernel<<<dim3(3 * DMODEL / 128, MTOT / 128), 256, 0, stream>>>(
      xb, wqkvb, b_qkv, qb, kb, vb);

  attn_kernel<<<dim3(SEQ / 128, NHEAD, BATCH), 256, 0, stream>>>(qb, kb, vb, attnb);

  gemm_out_kernel<<<dim3(DMODEL / 128, MTOT / 128), 256, 0, stream>>>(
      attnb, woutb, b_out, out);
}
